// Correlation_4174708211966
// MI455X (gfx1250) — matrix operand order repaired and checked by reading, not compiled, not run on hardware
//
#include <hip/hip_runtime.h>

// Correlation (FlowNet-style), PAD=MD=20, K=1, S1=1, S2=2, MUL=1
// inputs: x1,x2 (8,256,64,96) f32 ; output (8,441,64,96) f32
//
// Formulation: for fixed (b, y, oy) the 21 ox-displacements for all x are
// near-diagonal entries of the Gram matrix G = X1row @ X2row^T (96x256 each,
// K = channels).  Computed with v_wmma_f32_16x16x32_f16 in a band of 16x16
// tiles.  Invalid (oy out of range / ox out of range) entries stay 0 from a
// prior zero-fill pass.

typedef __attribute__((ext_vector_type(16))) _Float16 v16h;
typedef __attribute__((ext_vector_type(8)))  float    v8f;

#define CCH 256   // channels
#define HH  64
#define WW  96
#define ND  441   // 21*21 displacements
#define STR 264   // padded LDS stride (halfs) per x position: 256 + 8

__global__ void zero4_kernel(float4* __restrict__ out, int n4) {
    int i = blockIdx.x * blockDim.x + threadIdx.x;
    if (i < n4) out[i] = make_float4(0.f, 0.f, 0.f, 0.f);
}

__global__ __launch_bounds__(192) void corr_wmma_kernel(
    const float* __restrict__ x1, const float* __restrict__ x2,
    float* __restrict__ out)
{
    __shared__ __align__(16) _Float16 ldsA[WW * STR];  // x1 row y  [x][c] f16
    __shared__ __align__(16) _Float16 ldsB[WW * STR];  // x2 row y2 [x][c] f16

    const int by = blockIdx.x;          // 0 .. 8*64-1
    const int b = by / HH;
    const int y = by % HH;
    const int tid = threadIdx.x;

    const size_t chanStride = (size_t)HH * WW;           // per-channel stride
    const float* x1row = x1 + (size_t)b * CCH * chanStride + (size_t)y * WW;

    // ---- load + convert x1 row (96 x 256 f32 -> f16) ----
    for (int idx = tid; idx < CCH * WW; idx += 192) {
        int c = idx / WW;              // consecutive tids -> consecutive x (coalesced)
        int x = idx % WW;
        ldsA[x * STR + c] = (_Float16)x1row[c * chanStride + x];
    }

    const int wave = tid >> 5;          // 0..5  (wave32), uniform per wave
    const int lane = tid & 31;
    const int m    = lane & 15;         // A: M row / B: N col / D: N col
    const int half = lane >> 4;
    const int x0   = wave * 16;         // M-tile origin (x1 x-position)

    for (int j = 0; j < 21; ++j) {
        const int oy = (j - 10) * 2;
        const int y2 = y + oy;
        if (y2 < 0 || y2 >= HH) continue;          // block-uniform

        __syncthreads();   // prev compute done (also fences initial ldsA fill)

        // ---- load + convert x2 row y2 ----
        const float* x2row = x2 + (size_t)b * CCH * chanStride + (size_t)y2 * WW;
        for (int idx = tid; idx < CCH * WW; idx += 192) {
            int c = idx / WW;
            int x = idx % WW;
            ldsB[x * STR + c] = (_Float16)x2row[c * chanStride + x];
        }
        __syncthreads();

        // ---- band of N-tiles for this wave's M-tile ----
        int nlo = x0 - 32; if (nlo < 0)  nlo = 0;   // wave-uniform bounds
        int nhi = x0 + 32; if (nhi > 80) nhi = 80;  // tiles fully in [0,96)

        for (int xn0 = nlo; xn0 <= nhi; xn0 += 16) {
            v8f acc = {};
            #pragma unroll
            for (int kk = 0; kk < CCH; kk += 32) {
                v16h a, bm;
                // A 16x32 f16: lane M=m; vgpr0-3: K = kk+8*half+0..7,
                //                       vgpr4-7: K = kk+16+8*half+0..7
                const _Float16* pa = &ldsA[(x0 + m) * STR + kk + 8 * half];
                ((uint4*)&a)[0] = *(const uint4*)(pa);
                ((uint4*)&a)[1] = *(const uint4*)(pa + 16);
                // B 32x16 f16: lane N=m; vgpr0-3: K = kk+8*half+0..7, vgpr4-7: K = kk+16+8*half+0..7
                const _Float16* pb = &ldsB[(xn0 + m) * STR + kk + 8 * half];
                ((uint4*)&bm)[0] = *(const uint4*)(pb);
                ((uint4*)&bm)[1] = *(const uint4*)(pb + 16);
                acc = __builtin_amdgcn_wmma_f32_16x16x32_f16(
                          false, a, false, bm, (short)0, acc, false, false);
            }
            // ---- scatter the valid band entries ----
            // D layout: lane N = m, vgpr r -> M = r + 8*half
            #pragma unroll
            for (int r = 0; r < 8; ++r) {
                const int M   = r + 8 * half;
                const int x1p = x0 + M;         // x1 x-position
                const int x2p = xn0 + m;        // x2 x-position (in [0,96))
                const int ofs = x2p - x1p;      // ox displacement
                if (((ofs & 1) == 0) && ofs >= -20 && ofs <= 20) {
                    const int i = (ofs >> 1) + 10;
                    const int d = j * 21 + i;
                    out[(((size_t)b * ND + d) * HH + y) * WW + x1p] =
                        acc[r] * (1.0f / 256.0f);
                }
            }
        }
    }
}

extern "C" void kernel_launch(void* const* d_in, const int* in_sizes, int n_in,
                              void* d_out, int out_size, void* d_ws, size_t ws_size,
                              hipStream_t stream) {
    (void)in_sizes; (void)n_in; (void)d_ws; (void)ws_size;
    const float* x1 = (const float*)d_in[0];
    const float* x2 = (const float*)d_in[1];
    float* out = (float*)d_out;

    // zero-fill output (invalid displacements stay 0)
    int n4 = out_size >> 2;                       // out_size divisible by 4
    zero4_kernel<<<(n4 + 255) / 256, 256, 0, stream>>>((float4*)out, n4);

    // one block per (b, y); 6 waves, one 16-wide M-tile each
    corr_wmma_kernel<<<dim3(8 * HH), dim3(192), 0, stream>>>(x1, x2, out);
}
